// Mamba2Layer_16415365005654
// MI455X (gfx1250) — matrix operand order repaired and checked by reading, not compiled, not run on hardware
//
#include <hip/hip_runtime.h>
#include <hip/hip_bf16.h>

typedef __attribute__((ext_vector_type(16))) __bf16 v16bf;
typedef __attribute__((ext_vector_type(8)))  __bf16 v8bf;
typedef __attribute__((ext_vector_type(8)))  float  v8f;

#define HH   32
#define PP   64
#define NN   128
#define TT   2048
#define DIMD 2048
#define PROJD 12320       // 2*H*P + 2*H*N + H
#define BT   4096         // B*T

// ---------------------------------------------------------------------------
// Tiled bf16 WMMA GEMM:  Out[M,N] = A[M,K] * Bm[N,K]^T   (fp32 in, fp32 out)
// Block tile 256x128, K-step 32, 8 waves (4 along M x 2 along N).
// Each wave owns a 64x64 tile = 4x4 grid of v_wmma_f32_16x16x32_bf16:
// 16 WMMA per K-step vs 16 ds_load_b128 fragment loads (1:1), 4x frag reuse.
// ---------------------------------------------------------------------------
#define LDSK 40   // padded bf16 k-stride: 80 bytes/row -> 16B aligned, bank-spread

__global__ __launch_bounds__(256, 1) void gemm_bf16_wmma(
    const float* __restrict__ A,   // M x K   (M % 256 == 0)
    const float* __restrict__ Bm,  // N x K (row-major over K)
    float* __restrict__ Out,       // M x N
    int M, int Ncols, int K)
{
    (void)M;
    __shared__ __bf16 As[256 * LDSK];   // 20.0 KB
    __shared__ __bf16 Bs[128 * LDSK];   // 10.0 KB

    const int tid   = threadIdx.x;
    const int lane  = tid & 31;
    const int wid   = tid >> 5;
    const int waveM = wid & 3;     // 0..3 -> rows waveM*64
    const int waveN = wid >> 2;    // 0..1 -> cols waveN*64
    const int blockN = blockIdx.x * 128;
    const int blockM = blockIdx.y * 256;

    v8f acc[4][4];
#pragma unroll
    for (int i = 0; i < 4; ++i)
#pragma unroll
        for (int j = 0; j < 4; ++j) acc[i][j] = (v8f)0.0f;

    const int row  = tid >> 1;          // 0..127
    const int kseg = (tid & 1) * 16;    // 0 or 16

    for (int k0 = 0; k0 < K; k0 += 32) {
        // ---- stage A tile: 256 rows x 32 k (fp32 -> bf16) ----
#pragma unroll
        for (int rr = 0; rr < 2; ++rr) {
            const int r = row + rr * 128;
            const float* src = A + (size_t)(blockM + r) * K + k0 + kseg;
            v8bf lo, hi;
#pragma unroll
            for (int j = 0; j < 8; ++j) { lo[j] = (__bf16)src[j]; hi[j] = (__bf16)src[8 + j]; }
            *(v8bf*)&As[r * LDSK + kseg]     = lo;
            *(v8bf*)&As[r * LDSK + kseg + 8] = hi;
            if (k0 + 32 < K) __builtin_prefetch(src + 32, 0, 1);
        }
        // ---- stage B tile: 128 rows x 32 k (guard N edge) ----
        {
            const int nrow = blockN + row;
            v8bf lo = (v8bf)(__bf16)0.0f, hi = (v8bf)(__bf16)0.0f;
            if (nrow < Ncols) {
                const float* src = Bm + (size_t)nrow * K + k0 + kseg;
#pragma unroll
                for (int j = 0; j < 8; ++j) { lo[j] = (__bf16)src[j]; hi[j] = (__bf16)src[8 + j]; }
                if (k0 + 32 < K) __builtin_prefetch(src + 32, 0, 1);
            }
            *(v8bf*)&Bs[row * LDSK + kseg]     = lo;
            *(v8bf*)&Bs[row * LDSK + kseg + 8] = hi;
        }
        __syncthreads();

        // ---- fragments from LDS ----
        // A (16x32 bf16): lanes 0-15 hold K=[0..7,16..23], lanes 16-31 K=[8..15,24..31]
        const int khalf = (lane >= 16) ? 8 : 0;
        v16bf aF[4], bF[4];
#pragma unroll
        for (int tm = 0; tm < 4; ++tm) {
            const int m = waveM * 64 + tm * 16 + (lane & 15);
            v8bf lo = *(const v8bf*)&As[m * LDSK + khalf];
            v8bf hi = *(const v8bf*)&As[m * LDSK + 16 + khalf];
            aF[tm] = __builtin_shufflevector(lo, hi, 0,1,2,3,4,5,6,7,8,9,10,11,12,13,14,15);
        }
        // B (32x16 bf16): lanes 0-15 hold K=[0..7,16..23], lanes 16-31 K=[8..15,24..31]
        const int koff = (lane >= 16) ? 8 : 0;
#pragma unroll
        for (int tn = 0; tn < 4; ++tn) {
            const int n = waveN * 64 + tn * 16 + (lane & 15);
            v8bf lo = *(const v8bf*)&Bs[n * LDSK + koff];
            v8bf hi = *(const v8bf*)&Bs[n * LDSK + 16 + koff];
            bF[tn] = __builtin_shufflevector(lo, hi, 0,1,2,3,4,5,6,7,8,9,10,11,12,13,14,15);
        }
#pragma unroll
        for (int tm = 0; tm < 4; ++tm)
#pragma unroll
            for (int tn = 0; tn < 4; ++tn)
                acc[tm][tn] = __builtin_amdgcn_wmma_f32_16x16x32_bf16(
                    false, aF[tm], false, bF[tn], (short)0, acc[tm][tn], false, false);
        __syncthreads();
    }

    // ---- store: C/D layout VGPR r -> M = r + (lane>=16 ? 8 : 0), N = lane&15 ----
#pragma unroll
    for (int tm = 0; tm < 4; ++tm)
#pragma unroll
        for (int tn = 0; tn < 4; ++tn) {
            const int mBase = blockM + waveM * 64 + tm * 16 + ((lane >= 16) ? 8 : 0);
            const int col   = blockN + waveN * 64 + tn * 16 + (lane & 15);
            if (col < Ncols) {
#pragma unroll
                for (int r = 0; r < 8; ++r)
                    Out[(size_t)(mBase + r) * Ncols + col] = acc[tm][tn][r];
            }
        }
}

// ---------------------------------------------------------------------------
// dt = softplus(dt_raw + bias); decay = exp(dt * -exp(A_log))
// ---------------------------------------------------------------------------
__global__ __launch_bounds__(256) void dt_decay_kernel(
    const float* __restrict__ proj, const float* __restrict__ A_log,
    const float* __restrict__ dt_bias, float* __restrict__ dt_arr,
    float* __restrict__ dec_arr, int total)
{
    int i = blockIdx.x * 256 + threadIdx.x;
    if (i >= total) return;
    int h = i & (HH - 1);
    size_t bt = (size_t)(i >> 5);
    float raw = proj[bt * PROJD + (PROJD - HH) + h] + dt_bias[h];
    float dt  = (raw > 20.0f) ? raw : log1pf(__expf(raw));
    float Acoef = -__expf(A_log[h]);
    dt_arr[i]  = dt;
    dec_arr[i] = __expf(dt * Acoef);
}

// ---------------------------------------------------------------------------
// Causal conv (K=4, zero left-pad per batch) + SiLU; x-branch premultiplied by dt
// channel c in [0,10240): [0,2048)=x_ssm -> dtx, [2048,6144)=B -> Bc, rest=C -> Cc
// proj column = c (x) or c + 2048 (B/C, since z occupies [2048,4096)).
// ---------------------------------------------------------------------------
__global__ __launch_bounds__(256) void conv_silu_kernel(
    const float* __restrict__ proj, const float* __restrict__ dt_arr,
    const float* __restrict__ wx, const float* __restrict__ wb,
    const float* __restrict__ wc, float* __restrict__ dtx,
    float* __restrict__ Bc, float* __restrict__ Cc, long total)
{
    long i = (long)blockIdx.x * 256 + threadIdx.x;
    if (i >= total) return;
    const int  c  = (int)(i % 10240);
    const long bt = i / 10240;
    const int  t  = (int)(bt & (TT - 1));
    const long bbase = bt - t;   // b*T

    int col;
    const float* w;
    if (c < 2048)      { col = c;        w = wx + (size_t)c * 4; }
    else if (c < 6144) { col = c + 2048; w = wb + (size_t)(c - 2048) * 4; }
    else               { col = c + 2048; w = wc + (size_t)(c - 6144) * 4; }

    float accv = 0.0f;
#pragma unroll
    for (int k = 0; k < 4; ++k) {
        int tt = t - 3 + k;
        if (tt >= 0) accv += proj[(size_t)(bbase + tt) * PROJD + col] * w[k];
    }
    float sv = accv / (1.0f + __expf(-accv));   // SiLU

    if (c < 2048)      dtx[bt * 2048 + c]        = dt_arr[bt * HH + (c >> 6)] * sv;
    else if (c < 6144) Bc[bt * 4096 + (c - 2048)] = sv;
    else               Cc[bt * 4096 + (c - 6144)] = sv;
}

// ---------------------------------------------------------------------------
// Sequential SSM scan: 1 workgroup per (b,h). Thread owns p = tid>>2 and a
// 32-wide n-strip; 8192-float state lives in VGPRs. Per step: rank-1 update +
// dot, quad/wave shuffle reductions, RMS-norm over P, SiLU(z) gate.
// ---------------------------------------------------------------------------
#define CHK 8
__global__ __launch_bounds__(256) void scan_kernel(
    const float* __restrict__ proj, const float* __restrict__ dtx,
    const float* __restrict__ Bc, const float* __restrict__ Cc,
    const float* __restrict__ dec, float* __restrict__ yg)
{
    const int b    = blockIdx.x >> 5;
    const int h    = blockIdx.x & (HH - 1);
    const int tid  = threadIdx.x;
    const int p    = tid >> 2;
    const int quad = tid & 3;
    const int n0   = quad * 32;

    __shared__ float s_dtx[CHK][PP];
    __shared__ float s_z[CHK][PP];
    __shared__ float s_B[CHK][NN];
    __shared__ float s_C[CHK][NN];
    __shared__ float s_dec[CHK];
    __shared__ float s_wsum[8];

    float hstate[32];
#pragma unroll
    for (int i = 0; i < 32; ++i) hstate[i] = 0.0f;

    for (int t0 = 0; t0 < TT; t0 += CHK) {
        // stage chunk
        for (int i = tid; i < CHK * PP; i += 256) {
            const int s = i >> 6, pp = i & (PP - 1);
            const size_t bt = (size_t)b * TT + t0 + s;
            s_dtx[s][pp] = dtx[bt * 2048 + h * PP + pp];
            s_z[s][pp]   = proj[bt * PROJD + 2048 + h * PP + pp];
        }
        for (int i = tid; i < CHK * NN; i += 256) {
            const int s = i >> 7, nn = i & (NN - 1);
            const size_t bt = (size_t)b * TT + t0 + s;
            s_B[s][nn] = Bc[bt * 4096 + h * NN + nn];
            s_C[s][nn] = Cc[bt * 4096 + h * NN + nn];
        }
        if (tid < CHK) s_dec[tid] = dec[((size_t)b * TT + t0 + tid) * HH + h];
        __syncthreads();

        for (int s = 0; s < CHK; ++s) {
            const float d = s_dec[s];
            const float a = s_dtx[s][p];
            float accv = 0.0f;
#pragma unroll
            for (int i = 0; i < 32; ++i) {
                const float bv = s_B[s][n0 + i];
                const float cv = s_C[s][n0 + i];
                hstate[i] = d * hstate[i] + a * bv;
                accv += hstate[i] * cv;
            }
            // reduce over the 4-lane quad (full y[p] in every quad lane)
            accv += __shfl_xor(accv, 1, 32);
            accv += __shfl_xor(accv, 2, 32);
            // wave-level sum of y^2 over this wave's 8 p-rows (x4 duplication)
            float sq = accv * accv;
            sq += __shfl_xor(sq, 4, 32);
            sq += __shfl_xor(sq, 8, 32);
            sq += __shfl_xor(sq, 16, 32);
            sq *= 0.25f;
            if ((tid & 31) == 0) s_wsum[tid >> 5] = sq;
            __syncthreads();
            float ssum = 0.0f;
#pragma unroll
            for (int w = 0; w < 8; ++w) ssum += s_wsum[w];
            const float rinv = rsqrtf(ssum * (1.0f / 64.0f) + 1e-6f);
            if (quad == 0) {
                const float zv = s_z[s][p];
                const float g  = zv / (1.0f + __expf(-zv));
                yg[((size_t)b * TT + t0 + s) * 2048 + h * PP + p] = accv * rinv * g;
            }
            __syncthreads();
        }
    }
}

// ---------------------------------------------------------------------------
extern "C" void kernel_launch(void* const* d_in, const int* in_sizes, int n_in,
                              void* d_out, int out_size, void* d_ws, size_t ws_size,
                              hipStream_t stream) {
    (void)in_sizes; (void)n_in; (void)out_size; (void)ws_size;
    const float* x       = (const float*)d_in[0];
    const float* Win     = (const float*)d_in[1];
    const float* Wout    = (const float*)d_in[2];
    const float* A_log   = (const float*)d_in[3];
    const float* dt_bias = (const float*)d_in[4];
    const float* wx      = (const float*)d_in[5];
    const float* wb      = (const float*)d_in[6];
    const float* wc      = (const float*)d_in[7];
    float* out = (float*)d_out;
    float* ws  = (float*)d_ws;

    float* proj    = ws;                                   // BT x 12320
    float* dt_arr  = proj    + (size_t)BT * PROJD;         // BT x 32
    float* dec_arr = dt_arr  + (size_t)BT * HH;            // BT x 32
    float* dtx     = dec_arr + (size_t)BT * HH;            // BT x 2048
    float* Bcw     = dtx     + (size_t)BT * 2048;          // BT x 4096
    float* Ccw     = Bcw     + (size_t)BT * 4096;          // BT x 4096
    float* yg      = Ccw     + (size_t)BT * 4096;          // BT x 2048

    dim3 blk(256);

    // proj = x @ Win^T : M=4096, N=12320, K=2048
    gemm_bf16_wmma<<<dim3((PROJD + 127) / 128, BT / 256), blk, 0, stream>>>(
        x, Win, proj, BT, PROJD, DIMD);

    dt_decay_kernel<<<(BT * HH + 255) / 256, blk, 0, stream>>>(
        proj, A_log, dt_bias, dt_arr, dec_arr, BT * HH);

    const long convTotal = (long)BT * 10240;
    conv_silu_kernel<<<(unsigned)((convTotal + 255) / 256), blk, 0, stream>>>(
        proj, dt_arr, wx, wb, wc, dtx, Bcw, Ccw, convTotal);

    scan_kernel<<<2 * HH, blk, 0, stream>>>(proj, dtx, Bcw, Ccw, dec_arr, yg);

    // out = y_gated @ Wout^T : M=4096, N=2048, K=2048
    gemm_bf16_wmma<<<dim3(DIMD / 128, BT / 256), blk, 0, stream>>>(
        yg, Wout, out, BT, DIMD, HH * PP);
}
